// CellGraphGIN_84172769067903
// MI455X (gfx1250) — compile-verified, matrix operand order checked by reading, not run on hardware
//
#include <hip/hip_runtime.h>
#include <hip/hip_bf16.h>

#define DEVINL static __device__ __forceinline__

typedef __attribute__((ext_vector_type(16))) __bf16 v16bf;
typedef __attribute__((ext_vector_type(8)))  float  v8f;

DEVINL unsigned short f32_bf16_rne(float f) {
  unsigned int u = __builtin_bit_cast(unsigned int, f);
  unsigned int r = u + 0x7FFFu + ((u >> 16) & 1u);
  return (unsigned short)(r >> 16);
}

// async global->LDS copy of 16 bytes (per lane), tracked by ASYNCcnt
DEVINL void async_copy_b128(void* lds_dst, const void* gsrc) {
  unsigned int lds = (unsigned int)(unsigned long long)(uintptr_t)lds_dst; // low 32 bits = LDS offset
  unsigned long long ga = (unsigned long long)(uintptr_t)gsrc;
  asm volatile("global_load_async_to_lds_b128 %0, %1, off"
               :: "v"(lds), "v"(ga) : "memory");
}
DEVINL void async_wait0() {
  asm volatile("s_wait_asynccnt 0x0" ::: "memory");
}

// ---------------- weight fp32 -> bf16, TRANSPOSED + padded: dst[n][k] ----------------
__global__ void wconvT_kernel(const float* __restrict__ src, unsigned short* __restrict__ dst,
                              int K, int Nn, int Kpad, int Npad) {
  int idx = blockIdx.x * blockDim.x + threadIdx.x;
  if (idx >= Kpad * Npad) return;
  int n = idx / Kpad, k = idx % Kpad;
  float v = (k < K && n < Nn) ? src[(size_t)k * Nn + n] : 0.f;
  dst[(size_t)n * Kpad + k] = f32_bf16_rne(v);
}

// ---------------- layer-0 init: agg = x zero-padded to stride 64 ----------------
__global__ void pad0_kernel(const float* __restrict__ x, float* __restrict__ agg, long total) {
  long i = blockIdx.x * (long)blockDim.x + threadIdx.x;
  if (i >= total) return;
  long r = i >> 6; int c = (int)(i & 63);
  agg[i] = (c < 50) ? x[r * 50 + c] : 0.f;
}

// ---------------- generic init: agg = h (float4 copy) ----------------
__global__ void copy4_kernel(const float4* __restrict__ in, float4* __restrict__ out, long n4) {
  long i = blockIdx.x * (long)blockDim.x + threadIdx.x;
  if (i < n4) out[i] = in[i];
}

// ---------------- edge scatter-add: agg[dst] += h[src] (one wave per edge) ----------------
__global__ void scatter_kernel(const float* __restrict__ h, const int* __restrict__ src,
                               const int* __restrict__ dst, float* __restrict__ agg,
                               int E, int D, int hs, int as) {
  long gid = blockIdx.x * (long)blockDim.x + threadIdx.x;
  int lane = (int)(gid & 31);
  long e = gid >> 5;
  if (e >= E) return;
  const float* hp = h + (size_t)src[e] * hs;
  float* ap = agg + (size_t)dst[e] * as;
  for (int f = lane; f < D; f += 32) unsafeAtomicAdd(&ap[f], hp[f]);
}

// ---------------- bf16 WMMA GEMM: Out = act(A @ W + bias) ----------------
// A: fp32 [M][K] (row stride == K), WT: bf16 TRANSPOSED [N][ldw] (ldw = padded K), Out: fp32 [M][ldc]
template<int BM, int BN, int WM, int WN, bool RELU>
__global__ __launch_bounds__(WM * WN * 32)
void gemm_bf16_kernel(const float* __restrict__ A, const unsigned short* __restrict__ WT,
                      const float* __restrict__ bias, float* __restrict__ Out,
                      int M, int K, int ldw, int ldc, int nstore) {
  constexpr int KS  = 32;       // k-step (bf16 16x16x32)
  constexpr int LDK = KS + 8;   // padded LDS row (conflict-free fragment loads)
  constexpr int NTH = WM * WN * 32;
  constexpr int NT  = BN / (16 * WN);   // 16-col tiles per wave

  __shared__ __bf16 lsA[BM][LDK];       // [row][k]
  __shared__ __bf16 lsB[BN][LDK];       // [col][k]  (K-contiguous per column)

  const int tid  = threadIdx.x;
  const int lane = tid & 31;
  const int wave = tid >> 5;
  const int wm   = wave % WM;
  const int wn   = wave / WM;
  const int rowBase = blockIdx.x * BM;
  const int colBase = blockIdx.y * BN;

  v8f acc[NT];
  v8f zero = {0.f, 0.f, 0.f, 0.f, 0.f, 0.f, 0.f, 0.f};
#pragma unroll
  for (int t = 0; t < NT; ++t) acc[t] = zero;

  for (int k0 = 0; k0 < K; k0 += KS) {
    // 1) kick off async B tile copy (bf16 weights, k-contiguous): overlaps the A staging below
    for (int c = tid; c < BN * 4; c += NTH) {
      int col = c >> 2;
      int kc  = (c & 3) << 3;       // 8 bf16 = 16 bytes
      async_copy_b128(&lsB[col][kc],
                      WT + (size_t)(colBase + col) * ldw + k0 + kc);
    }
    // 2) stage A tile: fp32 -> bf16 (hardware cvt), float4 per thread-chunk
    for (int c = tid; c < BM * (KS / 4); c += NTH) {
      int r  = c >> 3;
      int kc = (c & 7) << 2;
      float4 v = {0.f, 0.f, 0.f, 0.f};
      int row = rowBase + r;
      if (row < M) v = *(const float4*)(A + (size_t)row * K + k0 + kc);
      lsA[r][kc + 0] = (__bf16)v.x;
      lsA[r][kc + 1] = (__bf16)v.y;
      lsA[r][kc + 2] = (__bf16)v.z;
      lsA[r][kc + 3] = (__bf16)v.w;
    }
    async_wait0();
    __syncthreads();

    // fragment loads per ISA 16-bit layouts: lanes<16 hold K {0..7,16..23}, lanes>=16 hold K {8..15,24..31}
    const int half = (lane & 16) ? 8 : 0;
    v16bf af;
    {
      const __bf16* ap = &lsA[wm * 16 + (lane & 15)][0];
#pragma unroll
      for (int j = 0; j < 8; ++j) { af[j] = ap[half + j]; af[8 + j] = ap[half + 16 + j]; }
    }
#pragma unroll
    for (int t = 0; t < NT; ++t) {
      const __bf16* bp = &lsB[wn * (16 * NT) + t * 16 + (lane & 15)][0];
      v16bf bfr;
#pragma unroll
      for (int j = 0; j < 8; ++j) { bfr[j] = bp[half + j]; bfr[8 + j] = bp[half + 16 + j]; }
      acc[t] = __builtin_amdgcn_wmma_f32_16x16x32_bf16(
          false, af, false, bfr, (short)0, acc[t], false, false);
    }
    __syncthreads();
  }

  // epilogue: D layout — lanes<16: M=v,N=lane ; lanes>=16: M=8+v,N=lane-16
  const int mOff = (lane & 16) ? 8 : 0;
#pragma unroll
  for (int t = 0; t < NT; ++t) {
    int n = colBase + wn * (16 * NT) + t * 16 + (lane & 15);
#pragma unroll
    for (int v = 0; v < 8; ++v) {
      int m = rowBase + wm * 16 + mOff + v;
      if (m < M && n < nstore) {
        float r = acc[t][v] + bias[n];
        if (RELU) r = fmaxf(r, 0.f);
        Out[(size_t)m * ldc + n] = r;
      }
    }
  }
}

// ---------------- BN batch statistics ----------------
__global__ void zero_stats_kernel(float* p) { p[threadIdx.x] = 0.f; } // <<<1,512>>>

__global__ void bn_stats_kernel(const float* __restrict__ h, float* __restrict__ sum,
                                float* __restrict__ sq, int M) {
  int c  = threadIdx.x;                 // 256 columns, coalesced
  int r0 = blockIdx.x * 512;
  int r1 = min(r0 + 512, M);
  float s = 0.f, q = 0.f;
  for (int r = r0; r < r1; ++r) {
    float v = h[(size_t)r * 256 + c];
    s += v; q += v * v;
  }
  unsafeAtomicAdd(&sum[c], s);
  unsafeAtomicAdd(&sq[c], q);
}

__global__ void bn_finalize_kernel(const float* __restrict__ sum, const float* __restrict__ sq,
                                   const float* __restrict__ gamma, const float* __restrict__ beta,
                                   float* __restrict__ scale, float* __restrict__ shift, int M) {
  int c = threadIdx.x;
  float inv = 1.f / (float)M;
  float mu  = sum[c] * inv;
  float var = fmaxf(sq[c] * inv - mu * mu, 0.f);
  float sc  = gamma[c] * rsqrtf(var + 1e-5f);
  scale[c] = sc;
  shift[c] = beta[c] - mu * sc;
}

__global__ void bn_apply_kernel(const float4* __restrict__ h2, const float* __restrict__ scale,
                                const float* __restrict__ shift, float4* __restrict__ out, long n4) {
  long i = blockIdx.x * (long)blockDim.x + threadIdx.x;
  if (i >= n4) return;
  int c4 = (int)(i & 63);
  float4 v  = h2[i];
  float4 sc = ((const float4*)scale)[c4];
  float4 sh = ((const float4*)shift)[c4];
  float4 o;
  o.x = fmaxf(fmaf(v.x, sc.x, sh.x), 0.f);
  o.y = fmaxf(fmaf(v.y, sc.y, sh.y), 0.f);
  o.z = fmaxf(fmaf(v.z, sc.z, sh.z), 0.f);
  o.w = fmaxf(fmaf(v.w, sc.w, sh.w), 0.f);
  out[i] = o;
}

// ---------------- host orchestration ----------------
extern "C" void kernel_launch(void* const* d_in, const int* in_sizes, int n_in,
                              void* d_out, int out_size, void* d_ws, size_t ws_size,
                              hipStream_t stream) {
  const float* x  = (const float*)d_in[0];
  const int*   ei = (const int*)d_in[1];
  const int N = in_sizes[0] / 50;
  const int E = in_sizes[1] / 2;
  const int* esrc = ei;
  const int* edst = ei + E;

  const float* w1[3] = {(const float*)d_in[2],  (const float*)d_in[8],  (const float*)d_in[14]};
  const float* b1[3] = {(const float*)d_in[3],  (const float*)d_in[9],  (const float*)d_in[15]};
  const float* w2[3] = {(const float*)d_in[4],  (const float*)d_in[10], (const float*)d_in[16]};
  const float* b2[3] = {(const float*)d_in[5],  (const float*)d_in[11], (const float*)d_in[17]};
  const float* gm[3] = {(const float*)d_in[6],  (const float*)d_in[12], (const float*)d_in[18]};
  const float* bt[3] = {(const float*)d_in[7],  (const float*)d_in[13], (const float*)d_in[19]};
  const float* wc = (const float*)d_in[20];
  const float* bc = (const float*)d_in[21];

  char* ws = (char*)d_ws;
  const size_t SZ = (size_t)N * 256 * sizeof(float);
  float* bufA = (float*)ws;          // h / t
  float* bufB = (float*)(ws + SZ);   // agg / h2
  char*  wp   = ws + 2 * SZ;
  unsigned short* w1b[3];
  unsigned short* w2b[3];
  for (int i = 0; i < 3; ++i) w1b[i] = (unsigned short*)(wp + (size_t)i * 131072);
  for (int i = 0; i < 3; ++i) w2b[i] = (unsigned short*)(wp + (3 + (size_t)i) * 131072);
  unsigned short* wcb = (unsigned short*)(wp + 6 * 131072);
  float* s_sum   = (float*)(wp + 6 * 131072 + 16384);
  float* s_sq    = s_sum + 256;
  float* s_scale = s_sum + 512;
  float* s_shift = s_sum + 768;

  // convert weights to bf16, transposed [N][Kpad] (w1_0 -> [256][64], wc -> [32][256])
  wconvT_kernel<<<(64 * 256 + 255) / 256, 256, 0, stream>>>(w1[0], w1b[0], 50, 256, 64, 256);
  for (int i = 1; i < 3; ++i)
    wconvT_kernel<<<256, 256, 0, stream>>>(w1[i], w1b[i], 256, 256, 256, 256);
  for (int i = 0; i < 3; ++i)
    wconvT_kernel<<<256, 256, 0, stream>>>(w2[i], w2b[i], 256, 256, 256, 256);
  wconvT_kernel<<<32, 256, 0, stream>>>(wc, wcb, 256, 20, 256, 32);

  const long n4 = (long)N * 64;   // float4 count of an N x 256 fp32 buffer

  for (int i = 0; i < 3; ++i) {
    if (i == 0) {
      // agg = x (padded to K=64), then += x[src]
      pad0_kernel<<<(unsigned)((n4 + 255) / 256), 256, 0, stream>>>(x, bufB, (long)N * 64);
      scatter_kernel<<<(unsigned)(((long)E * 32 + 255) / 256), 256, 0, stream>>>(
          x, esrc, edst, bufB, E, 50, 50, 64);
      gemm_bf16_kernel<64, 128, 4, 2, true><<<dim3((N + 63) / 64, 2), 256, 0, stream>>>(
          bufB, w1b[0], b1[0], bufA, N, 64, /*ldw=*/64, 256, 256);
    } else {
      copy4_kernel<<<(unsigned)((n4 + 255) / 256), 256, 0, stream>>>(
          (const float4*)bufA, (float4*)bufB, n4);
      scatter_kernel<<<(unsigned)(((long)E * 32 + 255) / 256), 256, 0, stream>>>(
          bufA, esrc, edst, bufB, E, 256, 256, 256);
      gemm_bf16_kernel<64, 128, 4, 2, true><<<dim3((N + 63) / 64, 2), 256, 0, stream>>>(
          bufB, w1b[i], b1[i], bufA, N, 256, /*ldw=*/256, 256, 256);
    }
    // h2 = t @ w2 + b2
    gemm_bf16_kernel<64, 128, 4, 2, false><<<dim3((N + 63) / 64, 2), 256, 0, stream>>>(
        bufA, w2b[i], b2[i], bufB, N, 256, /*ldw=*/256, 256, 256);
    // batch-norm stats + apply + relu
    zero_stats_kernel<<<1, 512, 0, stream>>>(s_sum);
    bn_stats_kernel<<<(N + 511) / 512, 256, 0, stream>>>(bufB, s_sum, s_sq, N);
    bn_finalize_kernel<<<1, 256, 0, stream>>>(s_sum, s_sq, gm[i], bt[i], s_scale, s_shift, N);
    bn_apply_kernel<<<(unsigned)((n4 + 255) / 256), 256, 0, stream>>>(
        (const float4*)bufB, s_scale, s_shift, (float4*)bufA, n4);
  }

  // classifier: [N,256] @ [256,20] + bc  (WMMA, N-padded to 32, store 20 cols)
  gemm_bf16_kernel<128, 32, 8, 1, false><<<dim3((N + 127) / 128, 1), 256, 0, stream>>>(
      bufA, wcb, bc, (float*)d_out, N, 256, /*ldw=*/256, 20, 20);
}
